// MultiHeadSelfAttention_10144712753563
// MI455X (gfx1250) — matrix operand order repaired and checked by reading, not compiled, not run on hardware
//
#include <hip/hip_runtime.h>
#include <hip/hip_bf16.h>

// ---------------------------------------------------------------------------
// MI455X (gfx1250) multi-head self-attention, bf16 WMMA pipeline with
// async global->LDS staging (ASYNCcnt) and double-buffered tiles.
// ---------------------------------------------------------------------------

static constexpr int kB = 8;
static constexpr int kS = 1024;
static constexpr int kE = 1024;
static constexpr int kH = 16;
static constexpr int kD = 64;   // kE / kH

typedef __attribute__((ext_vector_type(16))) __bf16       v16bf;
typedef __attribute__((ext_vector_type(8)))  float        v8f;
typedef __attribute__((ext_vector_type(4)))  unsigned int u32x4;

union FragBF {
  v16bf v;
  u32x4 u[2];
};

__device__ __forceinline__ unsigned short f2bf(float f) {
  unsigned int u = __float_as_uint(f);
  u += 0x7FFFu + ((u >> 16) & 1u);   // round-to-nearest-even
  return (unsigned short)(u >> 16);
}
__device__ __forceinline__ float bf2f(unsigned short h) {
  return __uint_as_float((unsigned int)h << 16);
}

__device__ __forceinline__ v8f wmma_bf16(const FragBF& a, const FragBF& b, v8f c) {
  return __builtin_amdgcn_wmma_f32_16x16x32_bf16(
      false, a.v, false, b.v, (short)0, c, false, false);
}

// --- async global->LDS 16B copy, with synchronous fallback -----------------
#if defined(__AMDGCN__) && __has_builtin(__builtin_amdgcn_global_load_async_to_lds_b128)
#define HAVE_ASYNC_LDS 1
#else
#define HAVE_ASYNC_LDS 0
#endif

// Builtin parameter types (from the round-2 diagnostic): pointee is a
// GCC-style int4 vector; param0 in addrspace(1) (global), param1 addrspace(3).
typedef int i32x4g __attribute__((vector_size(16)));
typedef __attribute__((address_space(1))) i32x4g* as1_i32x4p;
typedef __attribute__((address_space(3))) i32x4g* as3_i32x4p;

__device__ __forceinline__ void copy16_g2l(const unsigned short* g,
                                           unsigned short* l) {
#if HAVE_ASYNC_LDS
  __builtin_amdgcn_global_load_async_to_lds_b128(
      (as1_i32x4p)(uintptr_t)g, (as3_i32x4p)(uintptr_t)l, 0, 0);
#else
  *reinterpret_cast<ushort4*>(l) = *reinterpret_cast<const ushort4*>(g);
#endif
}

__device__ __forceinline__ void async_wait_all() {
#if defined(__AMDGCN__)
#if __has_builtin(__builtin_amdgcn_s_wait_asynccnt)
  __builtin_amdgcn_s_wait_asynccnt(0);
#else
  asm volatile("s_wait_asynccnt 0x0" ::: "memory");
#endif
#endif
}

// ---------------------------------------------------------------------------
// f32 -> bf16 bulk convert (X): 4 elements / thread
// ---------------------------------------------------------------------------
__global__ __launch_bounds__(256) void cvt_f32_to_bf16_x4(
    const float* __restrict__ in, unsigned short* __restrict__ out, int n4) {
  int i = blockIdx.x * 256 + threadIdx.x;
  if (i >= n4) return;
  float4 f = reinterpret_cast<const float4*>(in)[i];
  ushort4 o;
  o.x = f2bf(f.x); o.y = f2bf(f.y); o.z = f2bf(f.z); o.w = f2bf(f.w);
  reinterpret_cast<ushort4*>(out)[i] = o;
}

// ---------------------------------------------------------------------------
// f32 -> bf16 transposing convert (weights): out[n][k] = bf16(in[k][n]).
// One-time 4 MB pass per weight; keeps the GEMM hot loop transpose-free.
// ---------------------------------------------------------------------------
__global__ __launch_bounds__(256) void cvt_f32_to_bf16_tr(
    const float* __restrict__ in, unsigned short* __restrict__ out, int R,
    int C) {
  int i = blockIdx.x * 256 + threadIdx.x;  // index over out = C*R elements
  if (i >= R * C) return;
  int n = i / R, r = i % R;                // out[n][r]
  out[i] = f2bf(in[(size_t)r * C + n]);
}

// ---------------------------------------------------------------------------
// bf16 GEMM: C[MxN] = A[MxK] * Bt[NxK]^T (+bias).
// 256 threads = 8 waves (4x2); block tile 128x128, BK=32, double-buffered
// async LDS staging. Each wave: 32x64 tile = 8 WMMAs / k-step.
// MODE 0: bf16 row-major. MODE 1: f32 + bias. MODE 2: bf16, per-head
// d-major transpose (for V): f = row*N+col -> Vt[bh][d][s2].
// ---------------------------------------------------------------------------
template <int MODE>
__global__ __launch_bounds__(256) void gemm_bf16_wmma(
    const unsigned short* __restrict__ A, const unsigned short* __restrict__ Bt,
    void* __restrict__ C, const float* __restrict__ bias, int M, int N, int K) {
  constexpr int BM = 128, BN = 128, BK = 32;
  __shared__ unsigned short sA[2][BM * BK];  // [m][k]
  __shared__ unsigned short sB[2][BN * BK];  // [n][k]

  const int tid  = threadIdx.x;
  const int lane = tid & 31;
  const int wave = tid >> 5;   // 0..7
  const int wr   = wave >> 1;  // 0..3 (32-row group)
  const int wc   = wave & 1;   // 0..1 (64-col group)
  const int lh   = lane >> 4;
  const int ln   = lane & 15;
  const int m0   = blockIdx.x * BM;
  const int n0   = blockIdx.y * BN;

  v8f acc[2][4];
#pragma unroll
  for (int i = 0; i < 2; ++i)
#pragma unroll
    for (int j = 0; j < 4; ++j)
#pragma unroll
      for (int v = 0; v < 8; ++v) acc[i][j][v] = 0.0f;

  // Stage one 128x32 bf16 tile (512 x 16B chunks, 2 per thread).
  auto stage = [&](const unsigned short* __restrict__ g, int gstride,
                   unsigned short* l) {
    for (int i = tid; i < 512; i += 256) {
      int r = i >> 2, c = (i & 3) << 3;
      copy16_g2l(g + (size_t)r * gstride + c, l + r * BK + c);
    }
  };

  stage(A + (size_t)m0 * K, K, sA[0]);
  stage(Bt + (size_t)n0 * K, K, sB[0]);
  async_wait_all();
  __syncthreads();

  const int steps = K / BK;
  for (int s = 0; s < steps; ++s) {
    const int cur = s & 1;
    if (s + 1 < steps) {  // issue async stage of next k-slab before compute
      stage(A + (size_t)m0 * K + (s + 1) * BK, K, sA[cur ^ 1]);
      stage(Bt + (size_t)n0 * K + (s + 1) * BK, K, sB[cur ^ 1]);
    }

    FragBF fa[2];
#pragma unroll
    for (int i = 0; i < 2; ++i) {
      int row = wr * 32 + i * 16 + ln;
      fa[i].u[0] = *(const u32x4*)(sA[cur] + row * BK + lh * 8);
      fa[i].u[1] = *(const u32x4*)(sA[cur] + row * BK + 16 + lh * 8);
    }
#pragma unroll
    for (int j = 0; j < 4; ++j) {
      FragBF fb;
      int col = wc * 64 + j * 16 + ln;
      fb.u[0] = *(const u32x4*)(sB[cur] + col * BK + lh * 8);
      fb.u[1] = *(const u32x4*)(sB[cur] + col * BK + 16 + lh * 8);
      acc[0][j] = wmma_bf16(fa[0], fb, acc[0][j]);
      acc[1][j] = wmma_bf16(fa[1], fb, acc[1][j]);
    }
    async_wait_all();
    __syncthreads();
  }

  // Epilogue: C/D layout row = v + 8*lh, col = ln.
#pragma unroll
  for (int i = 0; i < 2; ++i)
#pragma unroll
    for (int j = 0; j < 4; ++j)
#pragma unroll
      for (int v = 0; v < 8; ++v) {
        int row = m0 + wr * 32 + i * 16 + v + 8 * lh;
        int col = n0 + wc * 64 + j * 16 + ln;
        float x = acc[i][j][v];
        size_t f = (size_t)row * N + col;
        if (MODE == 0) {
          ((unsigned short*)C)[f] = f2bf(x);
        } else if (MODE == 1) {
          ((float*)C)[f] = x + (bias ? bias[col] : 0.0f);
        } else {  // MODE 2: per-head transpose Vt[bh][d][s2]
          size_t idx = (f & ~(size_t)0xFFFF) + (size_t)(f & 63) * kS +
                       ((f >> 6) & (kS - 1));
          ((unsigned short*)C)[idx] = f2bf(x);
        }
      }
}

// ---------------------------------------------------------------------------
// Flash attention per (b,h). Q,K contiguous SxD bf16 blocks at bh*S*D;
// V pre-transposed per head: Vt[bh][d][s]. Output written pre-transposed
// into (b, s, h*D+d) bf16. 128 threads = 4 waves; 64 q-rows per block;
// double-buffered async K/V staging; scale 1/32 applied to f32 scores.
// ---------------------------------------------------------------------------
__global__ __launch_bounds__(128) void flash_attn_wmma(
    const unsigned short* __restrict__ Q, const unsigned short* __restrict__ K,
    const unsigned short* __restrict__ Vt, unsigned short* __restrict__ O) {
  __shared__ unsigned short sQ[64 * 64];      // [qrow][d]
  __shared__ unsigned short sK[2][64 * 64];   // [key][d]
  __shared__ unsigned short sVt[2][64 * 64];  // [d][key]
  __shared__ unsigned short sP[64 * 64];      // [qrow][key] per-wave strips

  const int tid  = threadIdx.x;
  const int lane = tid & 31;
  const int wave = tid >> 5;  // 0..3
  const int lh   = lane >> 4;
  const int ln   = lane & 15;
  const int qblk = blockIdx.x;  // 0..15
  const int bh   = blockIdx.y;  // 0..127
  const int b    = bh >> 4, h = bh & 15;
  const size_t base = (size_t)bh * kS * kD;  // also = bh*kD*kS for Vt
  const float scale = 0.03125f;              // 1024^-0.5 (exact)

  // Stage Q slab + tile 0 of K/V (each 64x64 bf16 = 512 x 16B chunks).
  for (int i = tid; i < 512; i += 128) {
    int r = i >> 3, c = (i & 7) << 3;
    copy16_g2l(Q + base + (size_t)(qblk * 64 + r) * kD + c, sQ + r * 64 + c);
    copy16_g2l(K + base + (size_t)r * kD + c, sK[0] + r * 64 + c);
    copy16_g2l(Vt + base + (size_t)r * kS + c, sVt[0] + r * 64 + c);
  }
  async_wait_all();
  __syncthreads();

  // Persistent Q fragments (QK^T reduces over d=64 -> two K-steps of 32).
  FragBF fq[2];
  {
    int row = wave * 16 + ln;
#pragma unroll
    for (int ks = 0; ks < 2; ++ks) {
      fq[ks].u[0] = *(const u32x4*)(sQ + row * 64 + ks * 32 + lh * 8);
      fq[ks].u[1] = *(const u32x4*)(sQ + row * 64 + ks * 32 + 16 + lh * 8);
    }
  }

  float m[8], l[8];
  v8f accO[4];
#pragma unroll
  for (int v = 0; v < 8; ++v) { m[v] = -3.0e38f; l[v] = 0.0f; }
#pragma unroll
  for (int n = 0; n < 4; ++n)
#pragma unroll
    for (int v = 0; v < 8; ++v) accO[n][v] = 0.0f;

  for (int t = 0; t < kS / 64; ++t) {
    const int cur = t & 1;
    if (t + 1 < kS / 64) {  // async-prefetch next K/V tiles
      for (int i = tid; i < 512; i += 128) {
        int r = i >> 3, c = (i & 7) << 3;
        copy16_g2l(K + base + (size_t)((t + 1) * 64 + r) * kD + c,
                   sK[cur ^ 1] + r * 64 + c);
        copy16_g2l(Vt + base + (size_t)r * kS + (t + 1) * 64 + c,
                   sVt[cur ^ 1] + r * 64 + c);
      }
    }

    // Scores: this wave's 16 rows x 64 keys (4 N-tiles), then scale.
    v8f accS[4];
#pragma unroll
    for (int j = 0; j < 4; ++j) {
      v8f s;
#pragma unroll
      for (int v = 0; v < 8; ++v) s[v] = 0.0f;
      int key = j * 16 + ln;
#pragma unroll
      for (int ks = 0; ks < 2; ++ks) {
        FragBF fk;
        fk.u[0] = *(const u32x4*)(sK[cur] + key * 64 + ks * 32 + lh * 8);
        fk.u[1] = *(const u32x4*)(sK[cur] + key * 64 + ks * 32 + 16 + lh * 8);
        s = wmma_bf16(fq[ks], fk, s);
      }
#pragma unroll
      for (int v = 0; v < 8; ++v) s[v] *= scale;
      accS[j] = s;
    }

    // Online softmax; rows live as (VGPR v, half lh); reduce over 16 lanes.
#pragma unroll
    for (int v = 0; v < 8; ++v) {
      float mx = accS[0][v];
#pragma unroll
      for (int j = 1; j < 4; ++j) mx = fmaxf(mx, accS[j][v]);
#pragma unroll
      for (int msk = 1; msk < 16; msk <<= 1)
        mx = fmaxf(mx, __shfl_xor(mx, msk, 32));
      float mn    = fmaxf(m[v], mx);
      float alpha = __expf(m[v] - mn);
      float rs    = 0.0f;
#pragma unroll
      for (int j = 0; j < 4; ++j) {
        float p = __expf(accS[j][v] - mn);
        accS[j][v] = p;
        rs += p;
      }
#pragma unroll
      for (int msk = 1; msk < 16; msk <<= 1) rs += __shfl_xor(rs, msk, 32);
      l[v] = l[v] * alpha + rs;
      m[v] = mn;
#pragma unroll
      for (int n = 0; n < 4; ++n) accO[n][v] *= alpha;
    }

    // Stage P (bf16) through this wave's private LDS strip.
#pragma unroll
    for (int j = 0; j < 4; ++j)
#pragma unroll
      for (int v = 0; v < 8; ++v)
        sP[(wave * 16 + v + 8 * lh) * 64 + j * 16 + ln] = f2bf(accS[j][v]);
    asm volatile("s_wait_dscnt 0x0" ::: "memory");  // same-wave LDS RAW fence

    // O += P(16x64) * V(64x64): reduce over keys -> two K-steps of 32.
    int prow = wave * 16 + ln;
#pragma unroll
    for (int ks = 0; ks < 2; ++ks) {
      FragBF fp;
      fp.u[0] = *(const u32x4*)(sP + prow * 64 + ks * 32 + lh * 8);
      fp.u[1] = *(const u32x4*)(sP + prow * 64 + ks * 32 + 16 + lh * 8);
#pragma unroll
      for (int n = 0; n < 4; ++n) {
        FragBF fv;
        int d = n * 16 + ln;
        fv.u[0] = *(const u32x4*)(sVt[cur] + d * 64 + ks * 32 + lh * 8);
        fv.u[1] = *(const u32x4*)(sVt[cur] + d * 64 + ks * 32 + 16 + lh * 8);
        accO[n] = wmma_bf16(fp, fv, accO[n]);
      }
    }
    async_wait_all();
    __syncthreads();
  }

  // Normalize and store transposed: out[b][s][h*D + d], bf16.
#pragma unroll
  for (int n = 0; n < 4; ++n)
#pragma unroll
    for (int v = 0; v < 8; ++v) {
      int row = qblk * 64 + wave * 16 + v + 8 * lh;
      int col = h * kD + n * 16 + ln;
      float o = accO[n][v] / l[v];
      O[(size_t)(b * kS + row) * kE + col] = f2bf(o);
    }
}

// ---------------------------------------------------------------------------
// Launch: convert -> 3x projection GEMM -> flash attention -> output GEMM.
// ---------------------------------------------------------------------------
extern "C" void kernel_launch(void* const* d_in, const int* in_sizes, int n_in,
                              void* d_out, int out_size, void* d_ws,
                              size_t ws_size, hipStream_t stream) {
  (void)in_sizes; (void)n_in; (void)out_size; (void)ws_size;
  const float* X  = (const float*)d_in[0];
  const float* Wq = (const float*)d_in[1];
  const float* Wk = (const float*)d_in[2];
  const float* Wv = (const float*)d_in[3];
  const float* Wo = (const float*)d_in[4];
  const float* bo = (const float*)d_in[5];

  const size_t nX = (size_t)kB * kS * kE;  // 8,388,608
  const size_t nW = (size_t)kE * kE;       // 1,048,576

  unsigned short* ws  = (unsigned short*)d_ws;
  size_t off = 0;
  unsigned short* Xb  = ws + off; off += nX;
  unsigned short* Wqb = ws + off; off += nW;  // transposed [N][K]
  unsigned short* Wkb = ws + off; off += nW;
  unsigned short* Wvb = ws + off; off += nW;
  unsigned short* Wob = ws + off; off += nW;
  unsigned short* Qb  = ws + off; off += nX;
  unsigned short* Kb  = ws + off; off += nX;
  unsigned short* VtB = ws + off; off += nX;  // per-head d-major V
  unsigned short* Ab  = ws + off; off += nX;  // attention out, (b,s,h*D+d)

  {
    int n4 = (int)(nX / 4);
    cvt_f32_to_bf16_x4<<<(n4 + 255) / 256, 256, 0, stream>>>(X, Xb, n4);
    int nw = (int)nW;
    cvt_f32_to_bf16_tr<<<(nw + 255) / 256, 256, 0, stream>>>(Wq, Wqb, kE, kE);
    cvt_f32_to_bf16_tr<<<(nw + 255) / 256, 256, 0, stream>>>(Wk, Wkb, kE, kE);
    cvt_f32_to_bf16_tr<<<(nw + 255) / 256, 256, 0, stream>>>(Wv, Wvb, kE, kE);
    cvt_f32_to_bf16_tr<<<(nw + 255) / 256, 256, 0, stream>>>(Wo, Wob, kE, kE);
  }

  const int M = kB * kS, N = kE, Kdim = kE;
  dim3 gGemm(M / 128, N / 128);

  gemm_bf16_wmma<0><<<gGemm, 256, 0, stream>>>(Xb, Wqb, (void*)Qb, nullptr, M, N, Kdim);
  gemm_bf16_wmma<0><<<gGemm, 256, 0, stream>>>(Xb, Wkb, (void*)Kb, nullptr, M, N, Kdim);
  gemm_bf16_wmma<2><<<gGemm, 256, 0, stream>>>(Xb, Wvb, (void*)VtB, nullptr, M, N, Kdim);

  flash_attn_wmma<<<dim3(kS / 64, kB * kH), 128, 0, stream>>>(Qb, Kb, VtB, Ab);

  gemm_bf16_wmma<1><<<gGemm, 256, 0, stream>>>(Ab, Wob, d_out, bo, M, N, Kdim);
}
